// RNNClassifier_1941325218421
// MI455X (gfx1250) — matrix operand order repaired and checked by reading, not compiled, not run on hardware
//
#include <hip/hip_runtime.h>

// RNNClassifier for MI455X (gfx1250): bf16 WMMA GEMMs + batch-partitioned
// recurrent scan (no inter-workgroup sync: recurrence is batch-row-local).
// xp is stored [t][h][b] so the serial scan reads it with contiguous b128s.

#define BB 64      // batch
#define CC 64      // input channels
#define TT 1024    // timesteps
#define HH 512     // hidden

typedef __attribute__((ext_vector_type(16))) __bf16       v16bf;
typedef __attribute__((ext_vector_type(8)))  float        v8f;
typedef __attribute__((ext_vector_type(4)))  float        v4f;
typedef __attribute__((ext_vector_type(4)))  unsigned int v4u;

// ---- WMMA fragment loaders (ISA 7.12.2 layouts, wave32) -------------------
// A 16x32 bf16: lanes 0-15 row M=lane, K chunks {kb..kb+7, kb+16..kb+23};
//               lanes 16-31 row M=lane-16, K chunks {kb+8..15, kb+24..31}.
__device__ __forceinline__ v16bf load_fragA(const __bf16* __restrict__ base,
                                            int stride, int row0, int kb, int lane) {
  const int sub = lane & 15;
  const int hi  = lane >> 4;
  const __bf16* p = base + (size_t)(row0 + sub) * stride + kb + hi * 8;
  union { v16bf v; v4u q[2]; } u;
  u.q[0] = *(const v4u*)(p);
  u.q[1] = *(const v4u*)(p + 16);
  return u.v;
}

// B 32x16 bf16 (B[k][n] = W[n][k], W row-major [N][K]):
// lanes 0-15 col N=lane, K chunks {kb..kb+7, kb+16..kb+23}; lanes 16-31 col N=lane-16, K chunks {kb+8..15, kb+24..31}.
__device__ __forceinline__ v16bf load_fragB(const __bf16* __restrict__ base,
                                            int stride, int col0, int kb, int lane) {
  const __bf16* p = base + (size_t)(col0 + (lane & 15)) * stride + kb + (lane >> 4) * 8;
  union { v16bf v; v4u q[2]; } u;
  u.q[0] = *(const v4u*)(p);
  u.q[1] = *(const v4u*)(p + 16);
  return u.v;
}

// ---- fp32 -> bf16 conversion kernels --------------------------------------
__global__ void k_f2bf(const float* __restrict__ src, __bf16* __restrict__ dst, int n) {
  int i = blockIdx.x * 256 + threadIdx.x;
  if (i < n) dst[i] = (__bf16)src[i];
}

// x [B][C][T] fp32 -> xs [B][T][C] bf16 (transpose + convert)
__global__ void k_xpose(const float* __restrict__ x, __bf16* __restrict__ xs) {
  int i = blockIdx.x * 256 + threadIdx.x;
  if (i < BB * TT * CC) {
    int c = i & (CC - 1);
    int t = (i >> 6) & (TT - 1);
    int b = i >> 16;
    xs[i] = (__bf16)x[((size_t)b * CC + c) * TT + t];
  }
}

// ---- Input-projection GEMM: xp[t][h][b] = act[b*T+t][:] . W[h][:] + bias[h]
// Grid: (M/64, H/128), block 256 (8 waves). Wave tile = 16 rows x 64 cols.
__global__ void __launch_bounds__(256)
k_proj(const __bf16* __restrict__ act, const __bf16* __restrict__ W,
       const float* __restrict__ bias, float* __restrict__ xp, int K) {
  const int tid  = threadIdx.x;
  const int lane = tid & 31;
  const int wv   = tid >> 5;
  const int row0 = blockIdx.x * 64 + (wv & 3) * 16;
  const int col0 = blockIdx.y * 128 + (wv >> 2) * 64;

  v8f acc[4];
#pragma unroll
  for (int j = 0; j < 4; ++j)
#pragma unroll
    for (int i = 0; i < 8; ++i) acc[j][i] = 0.0f;

  for (int kb = 0; kb < K; kb += 32) {
    v16bf a = load_fragA(act, K, row0, kb, lane);
#pragma unroll
    for (int j = 0; j < 4; ++j) {
      v16bf bm = load_fragB(W, K, col0 + j * 16, kb, lane);
      acc[j] = __builtin_amdgcn_wmma_f32_16x16x32_bf16(
          false, a, false, bm, (short)0, acc[j], false, false);
    }
  }

  // C/D layout: VGPR i -> M=(lane>>4)*8+i, N=(lane&15)
  const int nl = lane & 15;
  const int mo = (lane >> 4) * 8;
#pragma unroll
  for (int j = 0; j < 4; ++j) {
    const int n  = col0 + j * 16 + nl;
    const float bi = bias[n];
#pragma unroll
    for (int i = 0; i < 8; ++i) {
      const int r = row0 + mo + i;     // r = b*T + t
      const int b = r >> 10;
      const int t = r & (TT - 1);
      xp[((size_t)t * HH + n) * BB + b] = acc[j][i] + bi;   // [t][h][b]
    }
  }
}

// ---- Recurrent scan: one WG per 16 batch rows; h kept in LDS (bf16).
// No inter-WG sync required: h_t[b,:] depends only on h_{t-1}[b,:].
__global__ void __launch_bounds__(256)
k_scan(const float* __restrict__ xp, const __bf16* __restrict__ whh,
       const float* __restrict__ bhh, const float* __restrict__ h0,
       __bf16* __restrict__ ys, float* __restrict__ hT) {
  __shared__ __align__(16) __bf16 hsh[16 * HH];   // 16 KB
  const int b0   = blockIdx.x * 16;
  const int tid  = threadIdx.x;
  const int lane = tid & 31;
  const int wv   = tid >> 5;
  const int n0   = wv * 64;                        // wave owns 64 output cols

  for (int i = tid; i < 16 * HH; i += 256) {
    int r = i >> 9, h = i & (HH - 1);
    hsh[i] = (__bf16)h0[(size_t)(b0 + r) * HH + h];
  }
  __syncthreads();

  const int nl = lane & 15;
  const int mo = (lane >> 4) * 8;

  for (int t = 0; t < TT; ++t) {
    v8f acc[4];
#pragma unroll
    for (int j = 0; j < 4; ++j)
#pragma unroll
      for (int i = 0; i < 8; ++i) acc[j][i] = 0.0f;

    const float* xpt = xp + (size_t)t * HH * BB;   // xp[t][:][:]
    if (t + 1 < TT)  // global_prefetch_b8: pull next step's xp slab toward L2/L0
      __builtin_prefetch(xp + (size_t)(t + 1) * HH * BB + tid * 128, 0, 1);

    for (int kb = 0; kb < HH; kb += 32) {
      v16bf a = load_fragA(hsh, HH, 0, kb, lane);
#pragma unroll
      for (int j = 0; j < 4; ++j) {
        v16bf bm = load_fragB(whh, HH, n0 + j * 16, kb, lane);
        acc[j] = __builtin_amdgcn_wmma_f32_16x16x32_bf16(
            false, a, false, bm, (short)0, acc[j], false, false);
      }
    }
    __syncthreads();   // every wave finished reading hsh for step t

#pragma unroll
    for (int j = 0; j < 4; ++j) {
      const int n  = n0 + j * 16 + nl;
      const float bn = bhh[n];
      // 8 consecutive batch rows of xp[t][n][:] -> two b128 loads
      const float* px = xpt + (size_t)n * BB + b0 + mo;
      v4f x0 = *(const v4f*)(px);
      v4f x1 = *(const v4f*)(px + 4);
#pragma unroll
      for (int i = 0; i < 8; ++i) {
        const int m = mo + i;                       // local batch row
        const float xv = (i < 4) ? x0[i & 3] : x1[i & 3];
        float v  = acc[j][i] + xv + bn;
        float hn = tanhf(v);
        hsh[m * HH + n] = (__bf16)hn;
        if (ys) ys[((size_t)(b0 + m) * TT + t) * HH + n] = (__bf16)hn;
        if (hT && t == TT - 1) hT[(size_t)(b0 + m) * HH + n] = hn;
      }
    }
    __syncthreads();   // new h visible before step t+1 reads it
  }
}

// ---- Head: out[b] = sum_h relu(hT[b,h]) * Wfc[h] + bfc ---------------------
__global__ void k_head(const float* __restrict__ hT, const float* __restrict__ Wfc,
                       const float* __restrict__ bfc, float* __restrict__ out) {
  int b = threadIdx.x;
  if (b < BB) {
    float s = bfc[0];
    for (int h = 0; h < HH; ++h)
      s += fmaxf(hT[(size_t)b * HH + h], 0.0f) * Wfc[h];
    out[b] = s;
  }
}

extern "C" void kernel_launch(void* const* d_in, const int* in_sizes, int n_in,
                              void* d_out, int out_size, void* d_ws, size_t ws_size,
                              hipStream_t stream) {
  (void)in_sizes; (void)n_in; (void)out_size; (void)ws_size;
  const float* x     = (const float*)d_in[0];
  const float* h0    = (const float*)d_in[1];
  const float* W_ih0 = (const float*)d_in[2];
  const float* W_ihR = (const float*)d_in[3];
  const float* W_hh  = (const float*)d_in[4];
  const float* b_ih  = (const float*)d_in[5];
  const float* b_hh  = (const float*)d_in[6];
  const float* Wfc   = (const float*)d_in[7];
  const float* bfc   = (const float*)d_in[8];
  float* out = (float*)d_out;

  char* p = (char*)d_ws;
  auto take = [&](size_t bytes) -> void* {
    void* r = (void*)p;
    p += (bytes + 255) & ~(size_t)255;
    return r;
  };
  __bf16* xs   = (__bf16*)take((size_t)BB * TT * CC * 2);
  __bf16* w0   = (__bf16*)take((size_t)HH * CC * 2);
  __bf16* w1   = (__bf16*)take((size_t)HH * HH * 2);
  __bf16* w2   = (__bf16*)take((size_t)HH * HH * 2);
  __bf16* u0   = (__bf16*)take((size_t)HH * HH * 2);
  __bf16* u1   = (__bf16*)take((size_t)HH * HH * 2);
  __bf16* u2   = (__bf16*)take((size_t)HH * HH * 2);
  float*  hT   = (float*) take((size_t)BB * HH * 4);
  float*  xp   = (float*) take((size_t)TT * BB * HH * 4);
  __bf16* actA = (__bf16*)take((size_t)BB * TT * HH * 2);
  __bf16* actB = (__bf16*)take((size_t)BB * TT * HH * 2);

  // 1) convert inputs/weights to bf16
  k_xpose<<<(BB * TT * CC + 255) / 256, 256, 0, stream>>>(x, xs);
  k_f2bf<<<(HH * CC + 255) / 256, 256, 0, stream>>>(W_ih0, w0, HH * CC);
  k_f2bf<<<(HH * HH + 255) / 256, 256, 0, stream>>>(W_ihR,           w1, HH * HH);
  k_f2bf<<<(HH * HH + 255) / 256, 256, 0, stream>>>(W_ihR + HH * HH, w2, HH * HH);
  k_f2bf<<<(HH * HH + 255) / 256, 256, 0, stream>>>(W_hh,                u0, HH * HH);
  k_f2bf<<<(HH * HH + 255) / 256, 256, 0, stream>>>(W_hh + 1 * HH * HH, u1, HH * HH);
  k_f2bf<<<(HH * HH + 255) / 256, 256, 0, stream>>>(W_hh + 2 * HH * HH, u2, HH * HH);

  dim3 pg(BB * TT / 64, HH / 128);

  // 2) layer 0
  k_proj<<<pg, 256, 0, stream>>>(xs, w0, b_ih, xp, CC);
  k_scan<<<4, 256, 0, stream>>>(xp, u0, b_hh, h0, actB, nullptr);
  // 3) layer 1
  k_proj<<<pg, 256, 0, stream>>>(actB, w1, b_ih + HH, xp, HH);
  k_scan<<<4, 256, 0, stream>>>(xp, u1, b_hh + HH, h0 + BB * HH, actA, nullptr);
  // 4) layer 2
  k_proj<<<pg, 256, 0, stream>>>(actA, w2, b_ih + 2 * HH, xp, HH);
  k_scan<<<4, 256, 0, stream>>>(xp, u2, b_hh + 2 * HH, h0 + 2 * BB * HH, nullptr, hT);

  // 5) FC head
  k_head<<<1, 64, 0, stream>>>(hT, Wfc, bfc, out);
}
